// downsample_module_39737037422725
// MI455X (gfx1250) — compile-verified, matrix operand order checked by reading, not run on hardware
//
#include <hip/hip_runtime.h>

typedef __attribute__((ext_vector_type(16))) _Float16 v16h;
typedef __attribute__((ext_vector_type(8)))  float    v8f;

#define T_STEPS 4
#define Bn   16
#define Cc   128
#define Hh   56
#define Ww   56
#define OC   256
#define WO   28
#define Pp   784      // 28*28 positions per batch image
#define MT   49       // 16-row M tiles per batch (784/16)
#define NWAVE 784     // total M tiles = 16*49
#define NT   16       // N tiles (256/16)
#define KQ   4        // K chunks of 32 (128/32)
#define NROWS 12544.0f

// ---------------------------------------------------------------------------
// Kernel 1: pack conv_w into WMMA B-fragment layout, hi/lo FP16 split.
// B[k][n] = W[n][k].  Per-lane layout (16-bit operand, wave32):
//   lane = (hi<<4)|col ; vgpr v: k0 = q*32 + (v>>2)*16 + hi*8 + (v&3)*2
// dword index = (((plane*16+nt)*4+q)*32+lane)*8 + v  == tid
// ---------------------------------------------------------------------------
__global__ void pack_w_kernel(const float* __restrict__ w,
                              _Float16* __restrict__ bfrag) {
    int tid   = blockIdx.x * 256 + threadIdx.x;   // 0..32767
    int v     = tid & 7;
    int lane  = (tid >> 3) & 31;
    int q     = (tid >> 8) & 3;
    int nt    = (tid >> 10) & 15;
    int plane = (tid >> 14) & 1;                  // 0 = hi, 1 = lo
    int col = lane & 15, hi = lane >> 4;
    int k0  = q * 32 + (v >> 2) * 16 + hi * 8 + (v & 3) * 2;
    int o   = nt * 16 + col;
    float w0 = w[o * Cc + k0];
    float w1 = w[o * Cc + k0 + 1];
    _Float16 h0 = (_Float16)w0, h1 = (_Float16)w1;
    _Float16 e0, e1;
    if (plane == 0) { e0 = h0; e1 = h1; }
    else            { e0 = (_Float16)(w0 - (float)h0);
                      e1 = (_Float16)(w1 - (float)h1); }
    bfrag[tid * 2]     = e0;
    bfrag[tid * 2 + 1] = e1;
}

// ---------------------------------------------------------------------------
// Kernel 2: fused LIF scan (4 steps) + [::2,::2] downsample + scatter into
// WMMA A-fragment layout (FP16 is exact for values k/4).
// Only the strided quarter of x is ever read.
// ---------------------------------------------------------------------------
__global__ void lif_pack_a_kernel(const float* __restrict__ x,
                                  _Float16* __restrict__ afrag) {
    int tid = blockIdx.x * 256 + threadIdx.x;     // 0..1605631
    int p = tid % Pp;
    int c = (tid / Pp) % Cc;
    int b = tid / (Pp * Cc);
    int ho = p / WO, wo = p % WO;
    long base = ((long)b * Cc + c) * (Hh * Ww) + (long)(2 * ho) * Ww + 2 * wo;
    const long tstride = (long)Bn * Cc * Hh * Ww;
    float v = 0.0f; int cnt = 0;
#pragma unroll
    for (int t = 0; t < T_STEPS; ++t) {
        v += x[base + t * tstride];
        if (v >= 1.0f) { ++cnt; v = 0.0f; }       // spike + reset
    }
    float xs = (float)cnt * 0.25f;                // mean over T, exact in fp16
    int mtile = p >> 4, row = p & 15;
    int q = c >> 5, r = c & 31;
    int group = r >> 4, r2 = r & 15;
    int hi = r2 >> 3, r3 = r2 & 7;
    int within = r3 >> 1, pair = r3 & 1;
    int lane  = (hi << 4) | row;
    int v_reg = (group << 2) | within;
    long dword = ((((long)(b * MT + mtile)) * KQ + q) * 32 + lane) * 8 + v_reg;
    afrag[dword * 2 + pair] = (_Float16)xs;
}

// ---------------------------------------------------------------------------
// Kernel 3: GEMM y[m][o] = sum_c A[m][c] * W[o][c] via v_wmma_f32_16x16x32_f16
// One wave per 16-row M tile; loops over 16 N tiles; per N tile:
//   4 K-chunks x (hi + lo weight plane) = 8 WMMAs into one f32 accumulator.
// ---------------------------------------------------------------------------
__global__ void gemm_wmma_kernel(const _Float16* __restrict__ afrag,
                                 const _Float16* __restrict__ bfrag,
                                 float* __restrict__ y) {
    int wave = blockIdx.x * 8 + (threadIdx.x >> 5);  // 0..783
    int lane = threadIdx.x & 31;
    int hi = lane >> 4, nl = lane & 15;

    v16h A[KQ];
#pragma unroll
    for (int q = 0; q < KQ; ++q)
        A[q] = *(const v16h*)(afrag + (((long)wave * KQ + q) * 32 + lane) * 16);

    for (int nt = 0; nt < NT; ++nt) {
        v8f acc = {};
#pragma unroll
        for (int q = 0; q < KQ; ++q) {
            const _Float16* bh_p = bfrag + ((((0 * 16 + nt) * 4 + q) * 32 + lane) * 16);
            const _Float16* bl_p = bfrag + ((((1 * 16 + nt) * 4 + q) * 32 + lane) * 16);
            __builtin_prefetch(bh_p + 4096, 0, 3);
            v16h bh = *(const v16h*)bh_p;
            v16h bl = *(const v16h*)bl_p;
            acc = __builtin_amdgcn_wmma_f32_16x16x32_f16(
                false, A[q], false, bh, (short)0, acc, false, false);
            acc = __builtin_amdgcn_wmma_f32_16x16x32_f16(
                false, A[q], false, bl, (short)0, acc, false, false);
        }
#pragma unroll
        for (int vv = 0; vv < 8; ++vv) {
            int m = wave * 16 + vv + 8 * hi;         // global row (b*784 + p)
            y[(long)m * OC + nt * 16 + nl] = acc[vv];
        }
    }
}

// ---------------------------------------------------------------------------
// Kernel 4: per-channel partial sum / sumsq over 49-row slabs (deterministic).
// ---------------------------------------------------------------------------
__global__ void reduce_partial_kernel(const float* __restrict__ y,
                                      float* __restrict__ ps,
                                      float* __restrict__ pq) {
    int k = blockIdx.x;          // 0..255 slabs of 49 rows
    int o = threadIdx.x;         // channel
    float s = 0.0f, ss = 0.0f;
    long base = (long)k * 49 * OC + o;
    for (int r = 0; r < 49; ++r) {
        float val = y[base + (long)r * OC];
        s += val; ss += val * val;
    }
    ps[k * OC + o] = s;
    pq[k * OC + o] = ss;
}

// ---------------------------------------------------------------------------
// Kernel 5: finalize BN stats -> scale/shift per channel.
// ---------------------------------------------------------------------------
__global__ void finalize_stats_kernel(const float* __restrict__ ps,
                                      const float* __restrict__ pq,
                                      const float* __restrict__ gamma,
                                      const float* __restrict__ beta,
                                      float* __restrict__ scale,
                                      float* __restrict__ shift) {
    int o = threadIdx.x;
    float s = 0.0f, ss = 0.0f;
    for (int k = 0; k < 256; ++k) { s += ps[k * OC + o]; ss += pq[k * OC + o]; }
    const float invN = 1.0f / NROWS;
    float mean = s * invN;
    float var  = ss * invN - mean * mean;          // population variance
    float inv  = rsqrtf(var + 1e-5f);
    float sc   = gamma[o] * inv;
    scale[o] = sc;
    shift[o] = beta[o] - mean * sc;
}

// ---------------------------------------------------------------------------
// Kernel 6: apply affine, write reference layout (1,16,256,28,28). Coalesced
// stores; strided y reads hit L2 (y = 12.8 MB << 192 MB L2).
// ---------------------------------------------------------------------------
__global__ void bn_apply_kernel(const float* __restrict__ y,
                                const float* __restrict__ scale,
                                const float* __restrict__ shift,
                                float* __restrict__ out) {
    int idx = blockIdx.x;        // b*256 + o
    int b = idx >> 8, o = idx & 255;
    float sc = scale[o], sh = shift[o];
    long ybase = (long)b * Pp * OC + o;
    long obase = ((long)b * OC + o) * (long)Pp;
    for (int p = threadIdx.x; p < Pp; p += 256)
        out[obase + p] = y[ybase + (long)p * OC] * sc + sh;
}

// ---------------------------------------------------------------------------
extern "C" void kernel_launch(void* const* d_in, const int* in_sizes, int n_in,
                              void* d_out, int out_size, void* d_ws, size_t ws_size,
                              hipStream_t stream) {
    const float* x      = (const float*)d_in[0];
    const float* conv_w = (const float*)d_in[1];
    const float* gamma  = (const float*)d_in[2];
    const float* beta   = (const float*)d_in[3];
    float* out = (float*)d_out;

    // workspace layout (bytes)
    char* ws = (char*)d_ws;
    float*    y     = (float*)ws;                                 // 12,845,056 B
    _Float16* afrag = (_Float16*)(ws + 12845056);                 //  3,211,264 B
    _Float16* bfrag = (_Float16*)(ws + 12845056 + 3211264);       //    131,072 B
    float*    ps    = (float*)(ws + 12845056 + 3211264 + 131072); //    262,144 B
    float*    pq    = ps + 65536;                                 //    262,144 B
    float*    scale = pq + 65536;
    float*    shift = scale + 256;

    pack_w_kernel<<<128, 256, 0, stream>>>(conv_w, bfrag);
    lif_pack_a_kernel<<<6272, 256, 0, stream>>>(x, afrag);        // 16*128*784 threads
    gemm_wmma_kernel<<<98, 256, 0, stream>>>(afrag, bfrag, y);    // 784 waves
    reduce_partial_kernel<<<256, 256, 0, stream>>>(y, ps, pq);
    finalize_stats_kernel<<<1, 256, 0, stream>>>(ps, pq, gamma, beta, scale, shift);
    bn_apply_kernel<<<4096, 256, 0, stream>>>(y, scale, shift, out);
}
